// Lstm_20151986552989
// MI455X (gfx1250) — matrix operand order repaired and checked by reading, not compiled, not run on hardware
//
#include <hip/hip_runtime.h>
#include <hip/hip_bf16.h>

#define VOCAB 256
#define BIGF 128
#define NNSZ 1024
#define TLEN 512
#define BATCH 128
#define INDIM 259
#define INPAD 288
#define EMB 512
#define XDIM 640   // EMB + BIGF

typedef __attribute__((ext_vector_type(16))) __bf16 v16bf;
typedef __attribute__((ext_vector_type(8)))  float  v8f;

union Frag32B { v16bf v; uint4 q[2]; };

// ---------------------------------------------------------------------------
// A-fragment loader (bf16 16x32 MxK per CDNA5 ISA 7.12.2): lanes 0-15 hold
// K {k0..k0+7, k0+16..k0+23}; lanes 16-31 hold K {k0+8..k0+15, k0+24..k0+31};
// row M = lane%16. Two contiguous 16B loads per lane.
// ---------------------------------------------------------------------------
__device__ inline v16bf load_a_frag(const __bf16* __restrict__ base, int stride,
                                    int row0, int k0, int lane) {
    int m = lane & 15;
    int g = lane >> 4;
    const __bf16* p = base + (size_t)(row0 + m) * stride + k0 + 8 * g;
    Frag32B f;
    f.q[0] = *(const uint4*)(p);
    f.q[1] = *(const uint4*)(p + 16);
    return f.v;
}

// B-fragment from LDS-staged tile. LDS layout: per column c (tile-local),
// 4 consecutive uint4 = the 32 bf16 K-run. lane: n = lane%16 (column),
// kg = lane/16 -> uint4 kg and kg+2 (K {8kg..8kg+7, 16+8kg..16+8kg+7}).
__device__ inline v16bf load_b_frag_lds(const uint4* __restrict__ lds,
                                        int cidx, int lane) {
    int n  = lane & 15;
    int kg = lane >> 4;
    const uint4* p = lds + (size_t)(cidx + n) * 4 + kg;
    Frag32B f;
    f.q[0] = p[0];
    f.q[1] = p[2];
    return f.v;
}

// CDNA5 async global->LDS DMA (ASYNCcnt-tracked), 16 bytes per lane.
// VDST = 32-bit LDS byte address (flat LDS addresses truncate to addr[31:0]).
__device__ inline void async_ld16(unsigned lds_off, const void* gptr) {
    asm volatile("global_load_async_to_lds_b128 %0, %1, off"
                 :: "v"(lds_off), "v"(gptr)
                 : "memory");
}
__device__ inline void wait_async0() {
    asm volatile("s_wait_asynccnt 0x0" ::: "memory");
}

__device__ inline float sigmf(float x) { return 1.0f / (1.0f + __expf(-x)); }

// ---------------------------------------------------------------------------
// Generic bf16 WMMA GEMM: out[M,N] = A[M,K] * Wt[N,K]^T (+bias).
// 256 threads = 8 waves; wave = 16 rows x 64 cols; grid = (M/128, N/64).
// B tiles DMA'd global->LDS (ASYNCcnt), double-buffered:
// one s_wait_asynccnt + one barrier per K-step.
// epi: 0 = bf16 store, 1 = relu -> bf16, 2 = f32 store
// ---------------------------------------------------------------------------
__global__ __launch_bounds__(256) void gemm_bf16_kernel(
    const __bf16* __restrict__ A, int lda,
    const __bf16* __restrict__ Wt, int ldw, int K,
    const float* __restrict__ bias,
    void* __restrict__ out, int ldo, int epi)
{
    __shared__ uint4 ldsB[2][64 * 4];   // 2 x 4KB

    int tid  = threadIdx.x;
    int lane = tid & 31;
    int wave = tid >> 5;
    int row0 = blockIdx.x * 128 + wave * 16;
    int col0 = blockIdx.y * 64;
    int nk   = K >> 5;

    // Staging: thread -> (column, 16B chunk): 64 cols x 4 chunks = 256 threads
    int scol = col0 + (tid >> 2);
    int part = tid & 3;
    const __bf16* wrow = Wt + (size_t)scol * ldw + part * 8;
    unsigned ldsbase = (unsigned)(size_t)(&ldsB[0][0]) + (unsigned)tid * 16u;

    auto stage = [&](int buf, int k0) {
        async_ld16(ldsbase + (unsigned)buf * (64u * 64u), wrow + k0);
    };

    v8f acc[4] = {};

    stage(0, 0);
    v16bf a_cur = load_a_frag(A, lda, row0, 0, lane);
    wait_async0();
    __syncthreads();

    for (int i = 0; i < nk; ++i) {
        int buf   = i & 1;
        bool more = (i + 1 < nk);
        v16bf a_nxt = a_cur;
        if (more) {                                  // DMA next K-step -> buf^1
            int k0n = (i + 1) << 5;
            stage(buf ^ 1, k0n);
            a_nxt = load_a_frag(A, lda, row0, k0n, lane);
        }
#pragma unroll
        for (int nt = 0; nt < 4; ++nt) {
            v16bf b = load_b_frag_lds(ldsB[buf], nt * 16, lane);
            acc[nt] = __builtin_amdgcn_wmma_f32_16x16x32_bf16(
                false, a_cur, false, b, (short)0, acc[nt], false, false);
        }
        if (more) wait_async0();        // own DMAs landed; barrier covers rest
        __syncthreads();
        a_cur = a_nxt;
    }

    // C/D layout: VGPR r, lane l -> M = r + 8*(l/16), N = l%16
    int n  = lane & 15;
    int mh = lane >> 4;
#pragma unroll
    for (int nt = 0; nt < 4; ++nt) {
        int col = col0 + nt * 16 + n;
        float bv = bias ? bias[col] : 0.0f;
#pragma unroll
        for (int r = 0; r < 8; ++r) {
            int row = row0 + r + 8 * mh;
            float v = acc[nt][r] + bv;
            if (epi == 1) v = fmaxf(v, 0.0f);
            if (epi == 2)
                ((float*)out)[(size_t)row * ldo + col] = v;
            else
                ((__bf16*)out)[(size_t)row * ldo + col] = (__bf16)v;
        }
    }
}

// ---------------------------------------------------------------------------
// Fused LSTM step: z = [A1|A2] @ W (4096 cols) + b, gate math, c/h update.
// WG owns a 64-wide gate-column block across all 4 gates (256 z-columns),
// so the cell update is register-resident. grid = 16 WGs, 256 threads.
// B tiles DMA'd global->LDS with GLOBAL_LOAD_ASYNC_TO_LDS_B128 (ASYNCcnt),
// double-buffered: one s_wait_asynccnt + one barrier per K-step.
// ---------------------------------------------------------------------------
__global__ __launch_bounds__(256) void lstm_step_kernel(
    const __bf16* __restrict__ A1, int lda1, int K1,
    const __bf16* __restrict__ A2, int lda2, int K2,
    const __bf16* __restrict__ Wt, int ldw,          // [4096][K1+K2] N-major
    const float* __restrict__ bias,                  // [4096]
    float* __restrict__ c,                           // [128][1024]
    __bf16* __restrict__ hout)                       // [128][1024]
{
    __shared__ uint4 ldsB[2][256 * 4];  // 2 x 16KB: 256 cols x 32 K bf16

    int tid  = threadIdx.x;
    int lane = tid & 31;
    int wave = tid >> 5;
    int row0 = wave * 16;               // M = 128 = 8 waves x 16 rows
    int col0 = blockIdx.x * 64;
    int K    = K1 + K2;
    int nk   = K >> 5;

    // Staging column for this thread: gate = tid/64, col-in-block = tid%64
    int scol = (tid >> 6) * NNSZ + col0 + (tid & 63);
    const __bf16* wrow = Wt + (size_t)scol * ldw;
    unsigned ldsbase = (unsigned)(size_t)(&ldsB[0][0]) + (unsigned)tid * 64u;

    // Async-DMA this thread's 64-byte column run for K-step k0 into buffer buf.
    auto stage = [&](int buf, int k0) {
        const __bf16* g = wrow + k0;
        unsigned o = ldsbase + (unsigned)buf * (256u * 64u);
        async_ld16(o,      g);
        async_ld16(o + 16, g + 8);
        async_ld16(o + 32, g + 16);
        async_ld16(o + 48, g + 24);
    };
    auto loadA = [&](int k0) {
        return (k0 < K1) ? load_a_frag(A1, lda1, row0, k0, lane)
                         : load_a_frag(A2, lda2, row0, k0 - K1, lane);
    };

    v8f acc[4][4] = {};                 // [gate][ntile]

    stage(0, 0);
    v16bf a_cur = loadA(0);
    wait_async0();
    __syncthreads();

    for (int i = 0; i < nk; ++i) {
        int buf   = i & 1;
        bool more = (i + 1 < nk);
        v16bf a_nxt = a_cur;
        if (more) {                                  // DMA next K-step -> buf^1
            int k0n = (i + 1) << 5;
            stage(buf ^ 1, k0n);
            a_nxt = loadA(k0n);
        }
#pragma unroll
        for (int g = 0; g < 4; ++g) {
#pragma unroll
            for (int nt = 0; nt < 4; ++nt) {
                v16bf b = load_b_frag_lds(ldsB[buf], g * 64 + nt * 16, lane);
                acc[g][nt] = __builtin_amdgcn_wmma_f32_16x16x32_bf16(
                    false, a_cur, false, b, (short)0, acc[g][nt], false, false);
            }
        }
        if (more) wait_async0();        // own DMAs landed; barrier covers rest
        __syncthreads();
        a_cur = a_nxt;
    }

    int n  = lane & 15;
    int mh = lane >> 4;
#pragma unroll
    for (int nt = 0; nt < 4; ++nt) {
        int col = col0 + nt * 16 + n;
        float bf_ = bias[col];
        float bi_ = bias[NNSZ + col];
        float bo_ = bias[2 * NNSZ + col];
        float bg_ = bias[3 * NNSZ + col];
#pragma unroll
        for (int r = 0; r < 8; ++r) {
            int row = row0 + r + 8 * mh;
            size_t idx = (size_t)row * NNSZ + col;
            float gf = sigmf(acc[0][nt][r] + bf_);
            float gi = sigmf(acc[1][nt][r] + bi_);
            float go = sigmf(acc[2][nt][r] + bo_);
            float gg = tanhf(acc[3][nt][r] + bg_);
            float cn = gf * c[idx] + gi * gg;
            c[idx] = cn;
            hout[idx] = (__bf16)(go * tanhf(cn));
        }
    }
}

// ---------------------------------------------------------------------------
// Weight convert + transpose: W[K][N] f32 -> Wt[N][Kpad] bf16 (zero-padded K)
// ---------------------------------------------------------------------------
__global__ void convert_transpose_kernel(const float* __restrict__ W,
                                         __bf16* __restrict__ Wt,
                                         int K, int N, int Kpad)
{
    int idx = blockIdx.x * blockDim.x + threadIdx.x;
    if (idx >= N * Kpad) return;
    int nn = idx / Kpad;
    int k  = idx % Kpad;
    Wt[idx] = (k < K) ? (__bf16)W[(size_t)k * N + nn] : (__bf16)0.0f;
}

// Pack inputs: fp32 (TB,259) -> bf16 (TB,288) zero-padded; also write the
// "big" speaker-flag block of x_all (cols 512..639 = +/-1).
__global__ void pack_inputs_kernel(const float* __restrict__ in,
                                   __bf16* __restrict__ xpad,
                                   __bf16* __restrict__ x_all, int TB)
{
    int idx = blockIdx.x * blockDim.x + threadIdx.x;
    if (idx >= TB * INPAD) return;
    int tb = idx / INPAD;
    int k  = idx % INPAD;
    const float* row = in + (size_t)tb * INDIM;
    xpad[idx] = (k < INDIM) ? (__bf16)row[k] : (__bf16)0.0f;
    if (k < BIGF) {
        float comp = (row[VOCAB] == 1.0f && row[VOCAB + 1] == 0.0f) ? 1.0f : -1.0f;
        x_all[(size_t)tb * XDIM + EMB + k] = (__bf16)comp;
    }
}

// Zero recurrent state (c0, c1, h0 ping-buffer 0, h1_all step-0 block).
__global__ void init_state_kernel(float* __restrict__ c0, float* __restrict__ c1,
                                  __bf16* __restrict__ h0a, __bf16* __restrict__ h1z,
                                  int n)
{
    int i = blockIdx.x * blockDim.x + threadIdx.x;
    if (i >= n) return;
    c0[i] = 0.0f; c1[i] = 0.0f;
    h0a[i] = (__bf16)0.0f; h1z[i] = (__bf16)0.0f;
}

// ---------------------------------------------------------------------------
extern "C" void kernel_launch(void* const* d_in, const int* in_sizes, int n_in,
                              void* d_out, int out_size, void* d_ws, size_t ws_size,
                              hipStream_t stream)
{
    const float* inputs  = (const float*)d_in[0];  // (T,B,259)
    const float* emb_W   = (const float*)d_in[1];  // (259,512)
    const float* lstm_W0 = (const float*)d_in[2];  // (1664,4096)
    const float* lstm_b0 = (const float*)d_in[3];  // (4096)
    const float* lstm_W1 = (const float*)d_in[4];  // (2048,4096)
    const float* lstm_b1 = (const float*)d_in[5];  // (4096)
    const float* out_W0  = (const float*)d_in[6];  // (1024,1024)
    const float* out_b0  = (const float*)d_in[7];  // (1024)
    const float* out_W1  = (const float*)d_in[8];  // (1024,256)
    const float* out_b1  = (const float*)d_in[9];  // (256)
    float* logits = (float*)d_out;                 // (T,B,256)

    const int TB = TLEN * BATCH;                   // 65536
    const size_t ALIGN = 256;
    char* p = (char*)d_ws;
    auto alloc = [&](size_t bytes) {
        char* r = p;
        p += (bytes + ALIGN - 1) & ~(ALIGN - 1);
        return r;
    };
    __bf16* Wt0   = (__bf16*)alloc((size_t)4096 * 1664 * 2);
    __bf16* Wt1   = (__bf16*)alloc((size_t)4096 * 2048 * 2);
    __bf16* WtO0  = (__bf16*)alloc((size_t)1024 * 1024 * 2);
    __bf16* WtO1  = (__bf16*)alloc((size_t)256  * 1024 * 2);
    __bf16* WtE   = (__bf16*)alloc((size_t)EMB  * INPAD * 2);
    __bf16* xpad  = (__bf16*)alloc((size_t)TB * INPAD * 2);
    __bf16* x_all = (__bf16*)alloc((size_t)TB * XDIM * 2);
    __bf16* h0p   = (__bf16*)alloc((size_t)2 * BATCH * NNSZ * 2);   // ping-pong
    float*  c0    = (float*) alloc((size_t)BATCH * NNSZ * 4);
    float*  c1    = (float*) alloc((size_t)BATCH * NNSZ * 4);
    __bf16* h1all = (__bf16*)alloc((size_t)(TLEN + 1) * BATCH * NNSZ * 2);
    __bf16* hsall = (__bf16*)alloc((size_t)TB * NNSZ * 2);
    (void)ws_size; (void)n_in; (void)in_sizes; (void)out_size;

    // ---- Prologue: weight transpose/convert, input pack, state init ----
    auto t256 = [](size_t n) { return dim3((unsigned)((n + 255) / 256)); };
    convert_transpose_kernel<<<t256((size_t)4096 * 1664), 256, 0, stream>>>(lstm_W0, Wt0, 1664, 4096, 1664);
    convert_transpose_kernel<<<t256((size_t)4096 * 2048), 256, 0, stream>>>(lstm_W1, Wt1, 2048, 4096, 2048);
    convert_transpose_kernel<<<t256((size_t)1024 * 1024), 256, 0, stream>>>(out_W0, WtO0, 1024, 1024, 1024);
    convert_transpose_kernel<<<t256((size_t)256  * 1024), 256, 0, stream>>>(out_W1, WtO1, 1024, 256, 1024);
    convert_transpose_kernel<<<t256((size_t)EMB * INPAD), 256, 0, stream>>>(emb_W, WtE, INDIM, EMB, INPAD);
    pack_inputs_kernel<<<t256((size_t)TB * INPAD), 256, 0, stream>>>(inputs, xpad, x_all, TB);
    init_state_kernel<<<t256((size_t)BATCH * NNSZ), 256, 0, stream>>>(c0, c1, h0p, h1all, BATCH * NNSZ);

    // ---- Batched embedding GEMM: x_all[:, :512] = xpad @ emb_W (bf16) ----
    gemm_bf16_kernel<<<dim3(TB / 128, EMB / 64), 256, 0, stream>>>(
        xpad, INPAD, WtE, INPAD, INPAD, nullptr, x_all, XDIM, 0);

    // ---- Sequential scan: 2 fused LSTM kernels per step ----
    for (int t = 0; t < TLEN; ++t) {
        const __bf16* h0_prev = h0p + (size_t)(t & 1) * BATCH * NNSZ;
        __bf16*       h0_next = h0p + (size_t)((t + 1) & 1) * BATCH * NNSZ;
        const __bf16* x_t = x_all + (size_t)t * BATCH * XDIM;
        lstm_step_kernel<<<dim3(NNSZ / 64), 256, 0, stream>>>(
            x_t, XDIM, XDIM, h0_prev, NNSZ, NNSZ,
            Wt0, 1664, lstm_b0, c0, h0_next);
        const __bf16* h1_prev = h1all + (size_t)t * BATCH * NNSZ;
        __bf16*       h1_next = h1all + (size_t)(t + 1) * BATCH * NNSZ;
        lstm_step_kernel<<<dim3(NNSZ / 64), 256, 0, stream>>>(
            h0_next, NNSZ, NNSZ, h1_prev, NNSZ, NNSZ,
            Wt1, 2048, lstm_b1, c1, h1_next);
    }

    // ---- Batched output head over all T ----
    gemm_bf16_kernel<<<dim3(TB / 128, NNSZ / 64), 256, 0, stream>>>(
        h1all + (size_t)BATCH * NNSZ, NNSZ, WtO0, 1024, 1024, out_b0, hsall, NNSZ, 1);
    gemm_bf16_kernel<<<dim3(TB / 128, VOCAB / 64), 256, 0, stream>>>(
        hsall, NNSZ, WtO1, 1024, 1024, out_b1, logits, VOCAB, 2);
}
